// SelfAttention_18468359373398
// MI455X (gfx1250) — matrix operand order repaired and checked by reading, not compiled, not run on hardware
//
#include <hip/hip_runtime.h>

typedef __attribute__((ext_vector_type(16))) __bf16 v16bf;
typedef __attribute__((ext_vector_type(8)))  float  v8f;

union U16x16 {
    uint4 q[2];
    v16bf b;
    unsigned short s[16];
};

#define SEQ 2048
#define DMODEL 512
#define DH 64
#define NHEADS 8
#define BATCH 2

__device__ __forceinline__ unsigned short f2bf(float f) {
    union { float f; unsigned u; } v; v.f = f;
    unsigned u = v.u;
    u += 0x7FFFu + ((u >> 16) & 1u);   // round-to-nearest-even
    return (unsigned short)(u >> 16);
}

// ---------------------------------------------------------------- converts
__global__ void cvt_f32_bf16(const float4* __restrict__ src,
                             ushort4* __restrict__ dst, int n4) {
    int i = blockIdx.x * blockDim.x + threadIdx.x;
    if (i < n4) {
        float4 v = src[i];
        ushort4 o;
        o.x = f2bf(v.x); o.y = f2bf(v.y); o.z = f2bf(v.z); o.w = f2bf(v.w);
        dst[i] = o;
    }
}

// src [rows][512] f32 -> dst [512][rows] bf16   (cols hardcoded to 512)
__global__ void cvt_transpose512(const float* __restrict__ src,
                                 unsigned short* __restrict__ dst,
                                 int rows) {
    int i = blockIdx.x * blockDim.x + threadIdx.x;
    if (i < rows * 512) {
        int r = i >> 9, c = i & 511;
        dst[(size_t)c * rows + r] = f2bf(src[i]);
    }
}

// ---------------------------------------------------------------- GEMM
// C[M,N] = A[M,KD](bf16) @ BT[KD,N](bf16) + bias[N]
// 64x128 block tile, 8 waves, each wave 32x32 (2x2 WMMA tiles).
// A tile double-buffered in LDS via async b128 copies; B direct from L2.
// MODE 0: f32 -> outF.  MODE 1: split columns into Q / Kt / V (bf16).
template <int N, int KD, int MODE>
__global__ void gemm_bf16(const unsigned short* __restrict__ A,
                          const unsigned short* __restrict__ BT,
                          const float* __restrict__ bias,
                          float* __restrict__ outF,
                          unsigned short* __restrict__ Qo,
                          unsigned short* __restrict__ Kto,
                          unsigned short* __restrict__ Vo) {
    __shared__ unsigned short lds_a[2][64 * 32];   // 2 x 4KB ping-pong

    const int tid  = threadIdx.x;
    const int lane = tid & 31;
    const int wv   = tid >> 5;
    const int m0   = blockIdx.x * 64;
    const int n0   = blockIdx.y * 128;
    const int mi   = wv >> 2;           // 0..1 -> 32-row half
    const int ni   = wv & 3;            // 0..3 -> 32-col tile

    const int arow = tid >> 2;          // 0..63
    const int acol = (tid & 3) * 8;     // halves (16B per thread)
    const unsigned myoff =
        (unsigned)(unsigned long long)(&lds_a[0][0]) + (unsigned)tid * 16u;
    const unsigned short* agp = A + (size_t)(m0 + arow) * KD + acol;

    constexpr int NK = KD / 32;
    v8f acc[2][2] = {};

    const unsigned short* abase =
        (const unsigned short*)&lds_a[0][0] +
        (mi * 32 + (lane & 15)) * 32 + ((lane >> 4) * 8);

    // preload k-tile 0 into buffer 0
    asm volatile("global_load_async_to_lds_b128 %0, %1, off"
                 :: "v"(myoff), "v"(agp) : "memory");

    auto compute = [&](int i) {
        const int k0 = i * 32;
        const unsigned short* ab = abase + (i & 1) * 2048;   // halves (4KB)
        U16x16 a0, a1;
        a0.q[0] = *(const uint4*)ab;            a0.q[1] = *(const uint4*)(ab + 16);
        a1.q[0] = *(const uint4*)(ab + 512);    a1.q[1] = *(const uint4*)(ab + 528);

        const unsigned short* bp = BT + (size_t)k0 * N + n0 + ni * 32 + (lane & 15);
        U16x16 b0, b1;
        #pragma unroll
        for (int s = 0; s < 16; ++s) {
            const int kk = (lane >> 4) * 8 + (s & 7) + (s >> 3) * 16;
            b0.s[s] = bp[(size_t)kk * N];
            b1.s[s] = bp[(size_t)kk * N + 16];
        }

        acc[0][0] = __builtin_amdgcn_wmma_f32_16x16x32_bf16(false, a0.b, false, b0.b,
                                                            (short)0, acc[0][0], false, false);
        acc[0][1] = __builtin_amdgcn_wmma_f32_16x16x32_bf16(false, a0.b, false, b1.b,
                                                            (short)0, acc[0][1], false, false);
        acc[1][0] = __builtin_amdgcn_wmma_f32_16x16x32_bf16(false, a1.b, false, b0.b,
                                                            (short)0, acc[1][0], false, false);
        acc[1][1] = __builtin_amdgcn_wmma_f32_16x16x32_bf16(false, a1.b, false, b1.b,
                                                            (short)0, acc[1][1], false, false);
    };

    // steady state: always issue next tile, wait for current, compute
    for (int i = 0; i < NK - 1; ++i) {
        const unsigned short* gp = agp + (i + 1) * 32;
        const unsigned dstoff = myoff + (unsigned)(((i + 1) & 1) * 4096);
        asm volatile("global_load_async_to_lds_b128 %0, %1, off"
                     :: "v"(dstoff), "v"(gp) : "memory");
        asm volatile("s_wait_asynccnt 0x1" ::: "memory");   // tile i landed
        __syncthreads();
        compute(i);
        __syncthreads();
    }
    // final tile
    asm volatile("s_wait_asynccnt 0x0" ::: "memory");
    __syncthreads();
    compute(NK - 1);

    const int cn   = lane & 15;
    const int mrow = (lane >> 4) * 8;
    #pragma unroll
    for (int t = 0; t < 2; ++t) {
        #pragma unroll
        for (int u = 0; u < 2; ++u) {
            const int gn = n0 + ni * 32 + u * 16 + cn;
            const float bv = bias[gn];
            if (MODE == 0) {
                #pragma unroll
                for (int r = 0; r < 8; ++r)
                    outF[(size_t)(m0 + mi * 32 + t * 16 + mrow + r) * N + gn] =
                        acc[t][u][r] + bv;
            } else {
                const int part   = gn / DMODEL;      // 0=Q 1=K 2=V
                const int within = gn % DMODEL;
                const int h = within / DH, d = within % DH;
                #pragma unroll
                for (int r = 0; r < 8; ++r) {
                    const int mg = m0 + mi * 32 + t * 16 + mrow + r;
                    const int bb = mg / SEQ, srow = mg % SEQ;
                    const size_t bh = (size_t)(bb * NHEADS + h);
                    const unsigned short val = f2bf(acc[t][u][r] + bv);
                    if (part == 0)      Qo [(bh * SEQ + srow) * DH + d] = val;
                    else if (part == 1) Kto[(bh * DH + d) * SEQ + srow] = val;
                    else                Vo [(bh * SEQ + srow) * DH + d] = val;
                }
            }
        }
    }
}

// ---------------------------------------------------------------- attention
// one block = one (b,h) and 16 queries; full 16x2048 score row kept in LDS.
// After softmax/top-k, sparsified weights are re-written into the upper half
// of each row's 8KB LDS slot as bf16, so phase 4 A-frags are two ds_load_b128.
__global__ void attn_kernel(const unsigned short* __restrict__ Q,
                            const unsigned short* __restrict__ Kt,
                            const unsigned short* __restrict__ V,
                            unsigned short* __restrict__ attnb) {
    extern __shared__ char smem_raw[];
    float* sc  = (float*)smem_raw;                     // [16][2048] f32, 128 KB
    unsigned short* wbf = (unsigned short*)smem_raw;   // bf16 view (row*4096+2048)
    float* red = sc + 16 * SEQ;                        // [16][64], 4 KB

    const int tid  = threadIdx.x;
    const int lane = tid & 31;
    const int wv   = tid >> 5;
    const int bh   = blockIdx.x >> 7;
    const int qt   = blockIdx.x & 127;
    const int q0   = qt * 16;

    const int mloc = lane & 15;
    const int kh8  = (lane >> 4) * 8;

    // Q A-fragments: 16x64 -> two 16x32 frags per wave
    const unsigned short* qrow = Q + (size_t)(bh * SEQ + q0 + mloc) * DH;
    U16x16 a0, a1;
    a0.q[0] = *(const uint4*)(qrow + kh8);
    a0.q[1] = *(const uint4*)(qrow + 16 + kh8);
    a1.q[0] = *(const uint4*)(qrow + 32 + kh8);
    a1.q[1] = *(const uint4*)(qrow + 48 + kh8);

    // ---- phase 1: scores = (Q K^T) * scale -> LDS (f32)
    #pragma unroll 2
    for (int kt = wv; kt < SEQ / 16; kt += 8) {
        const int n0 = kt * 16;
        const unsigned short* kp0 = Kt + (size_t)(bh * DH) * SEQ + n0 + (lane & 15);
        U16x16 b0, b1;
        #pragma unroll
        for (int s = 0; s < 16; ++s) {
            const int kk = kh8 + (s & 7) + (s >> 3) * 16;
            b0.s[s] = kp0[(size_t)kk * SEQ];
            b1.s[s] = kp0[(size_t)(kk + 32) * SEQ];
        }
        v8f c = {};
        c = __builtin_amdgcn_wmma_f32_16x16x32_bf16(false, a0.b, false, b0.b,
                                                    (short)0, c, false, false);
        c = __builtin_amdgcn_wmma_f32_16x16x32_bf16(false, a1.b, false, b1.b,
                                                    (short)0, c, false, false);
        const int cn   = n0 + (lane & 15);
        const int mrow = (lane >> 4) * 8;
        #pragma unroll
        for (int r = 0; r < 8; ++r)
            sc[(mrow + r) * SEQ + cn] = c[r] * 0.125f;   // 1/sqrt(64)
    }
    __syncthreads();

    // ---- phase 2: softmax + entropy + dynamic top-k (row cached in regs)
    for (int rr = 0; rr < 2; ++rr) {
        const int row = wv * 2 + rr;
        float* wrow = sc + (size_t)row * SEQ;
        float wl[64];
        #pragma unroll
        for (int j = 0; j < 64; ++j) wl[j] = wrow[j * 32 + lane];

        float mx = -1e30f;
        #pragma unroll
        for (int j = 0; j < 64; ++j) mx = fmaxf(mx, wl[j]);
        #pragma unroll
        for (int o = 16; o > 0; o >>= 1) mx = fmaxf(mx, __shfl_xor(mx, o, 32));

        float sum = 0.f;
        #pragma unroll
        for (int j = 0; j < 64; ++j) { wl[j] = __expf(wl[j] - mx); sum += wl[j]; }
        #pragma unroll
        for (int o = 16; o > 0; o >>= 1) sum += __shfl_xor(sum, o, 32);
        const float inv = 1.f / sum;

        float ent = 0.f;
        #pragma unroll
        for (int j = 0; j < 64; ++j) { wl[j] *= inv; ent -= wl[j] * __logf(wl[j] + 1e-8f); }
        #pragma unroll
        for (int o = 16; o > 0; o >>= 1) ent += __shfl_xor(ent, o, 32);

        int topk = (int)(32.f * (1.f - ent));
        topk = topk < 1 ? 1 : (topk > 32 ? 32 : topk);

        float lo = 0.f, hi = 1.f;
        for (int it = 0; it < 24; ++it) {
            const float mid = 0.5f * (lo + hi);
            int cnt = 0;
            #pragma unroll
            for (int j = 0; j < 64; ++j) cnt += (wl[j] >= mid) ? 1 : 0;
            #pragma unroll
            for (int o = 16; o > 0; o >>= 1) cnt += __shfl_xor(cnt, o, 32);
            if (cnt >= topk) lo = mid; else hi = mid;
        }
        const float thr = lo;

        float s2 = 0.f;
        #pragma unroll
        for (int j = 0; j < 64; ++j) { if (wl[j] >= thr) s2 += wl[j]; else wl[j] = 0.f; }
        #pragma unroll
        for (int o = 16; o > 0; o >>= 1) s2 += __shfl_xor(s2, o, 32);
        const float rn = 1.f / (s2 + 1e-8f);

        // write final sparse weights as bf16 into row's own upper half
        unsigned short* wrow_bf = wbf + row * 4096 + 2048;
        #pragma unroll
        for (int j = 0; j < 64; ++j) wrow_bf[j * 32 + lane] = f2bf(wl[j] * rn);
    }
    __syncthreads();

    // ---- phase 4: out16x64 = W[16,2048](bf16 LDS) @ V[2048,64]
    const int ni    = wv & 3;          // dh tile
    const int khalf = wv >> 2;         // key half (1024 each)
    v8f c = {};
    #pragma unroll 4
    for (int j = 0; j < 32; ++j) {
        const int k0 = khalf * 1024 + j * 32;
        const unsigned short* ap = wbf + mloc * 4096 + 2048 + k0 + kh8;
        U16x16 af;
        af.q[0] = *(const uint4*)ap;
        af.q[1] = *(const uint4*)(ap + 16);
        const unsigned short* vp = V + (size_t)(bh * SEQ + k0) * DH + ni * 16 + (lane & 15);
        U16x16 bf_;
        #pragma unroll
        for (int s = 0; s < 16; ++s)
            bf_.s[s] = vp[(size_t)(kh8 + (s & 7) + (s >> 3) * 16) * DH];
        c = __builtin_amdgcn_wmma_f32_16x16x32_bf16(false, af.b, false, bf_.b,
                                                    (short)0, c, false, false);
    }

    const int cn   = ni * 16 + (lane & 15);
    const int mrow = (lane >> 4) * 8;
    if (khalf == 1) {
        #pragma unroll
        for (int r = 0; r < 8; ++r) red[(mrow + r) * 64 + cn] = c[r];
    }
    __syncthreads();
    if (khalf == 0) {
        const int bb = bh >> 3, h = bh & 7;
        #pragma unroll
        for (int r = 0; r < 8; ++r) {
            const float v = c[r] + red[(mrow + r) * 64 + cn];
            const int srow = q0 + mrow + r;
            attnb[(size_t)(bb * SEQ + srow) * DMODEL + h * DH + cn] = f2bf(v);
        }
    }
}

// ---------------------------------------------------------------- launch
extern "C" void kernel_launch(void* const* d_in, const int* in_sizes, int n_in,
                              void* d_out, int out_size, void* d_ws, size_t ws_size,
                              hipStream_t stream) {
    const float* x     = (const float*)d_in[0];   // [2,2048,512]
    const float* w_in  = (const float*)d_in[1];   // [1536,512]
    const float* b_in  = (const float*)d_in[2];   // [1536]
    const float* w_out = (const float*)d_in[3];   // [512,512]
    const float* b_out = (const float*)d_in[4];   // [512]
    float* out = (float*)d_out;

    unsigned short* ws    = (unsigned short*)d_ws;
    unsigned short* xb    = ws;                    // 2097152
    unsigned short* winT  = xb    + 2097152;       // 786432  [512][1536]
    unsigned short* woutT = winT  + 786432;        // 262144  [512][512]
    unsigned short* Qb    = woutT + 262144;        // 2097152 [bh][s][d]
    unsigned short* Ktb   = Qb    + 2097152;       // 2097152 [bh][d][s]
    unsigned short* Vb    = Ktb   + 2097152;       // 2097152 [bh][s][d]
    unsigned short* attnb = Vb    + 2097152;       // 2097152 [b][s][D]

    cvt_f32_bf16 <<<(2097152 / 4 + 255) / 256, 256, 0, stream>>>(
        (const float4*)x, (ushort4*)xb, 2097152 / 4);
    cvt_transpose512<<<(786432 + 255) / 256, 256, 0, stream>>>(w_in,  winT, 1536);
    cvt_transpose512<<<(262144 + 255) / 256, 256, 0, stream>>>(w_out, woutT, 512);

    dim3 g1(4096 / 64, 1536 / 128);
    gemm_bf16<1536, 512, 1><<<g1, 256, 0, stream>>>(xb, winT, b_in,
                                                    nullptr, Qb, Ktb, Vb);

    attn_kernel<<<BATCH * NHEADS * (SEQ / 16), 256,
                  16 * SEQ * 4 + 16 * 64 * 4, stream>>>(Qb, Ktb, Vb, attnb);

    dim3 g2(4096 / 64, 512 / 128);
    gemm_bf16<512, 512, 0><<<g2, 256, 0, stream>>>(attnb, woutT, b_out,
                                                   out, nullptr, nullptr, nullptr);
}
